// Attention_79216376808199
// MI455X (gfx1250) — matrix operand order repaired and checked by reading, not compiled, not run on hardware
//
#include <hip/hip_runtime.h>
#include <hip/hip_bf16.h>

// ---------------------------------------------------------------------------
// CDNA5 (gfx1250) GQA attention forward, bf16 WMMA (v_wmma_f32_16x16x32_bf16)
// ---------------------------------------------------------------------------

typedef __bf16 bf16;
typedef __attribute__((ext_vector_type(16))) __bf16 bf16x16;
typedef __attribute__((ext_vector_type(8)))  float   floatx8;

struct alignas(16) f4 { float x, y, z, w; };

__device__ __forceinline__ floatx8 wmma_bf16(bf16x16 a, bf16x16 b, floatx8 c) {
  // 8 args: (neg_a, A, neg_b, B, c_mod, C, reuse_a, reuse_b)
  return __builtin_amdgcn_wmma_f32_16x16x32_bf16(false, a, false, b,
                                                 (short)0, c, false, false);
}

union FragU { bf16x16 v; f4 q[2]; };

// A fragment (16x32, 16-bit): lane = M%16 in each half; half 0 holds K
// [0..7] and [16..23], half 1 holds K [8..15] and [24..31]. (ISA 7.12.2)
__device__ __forceinline__ bf16x16 ld_frag_a(const bf16* base, int m0,
                                             int stride, int k0) {
  const int lane = threadIdx.x & 31;
  const bf16* p = base + (size_t)(m0 + (lane & 15)) * stride + k0 + (lane >> 4) * 8;
  FragU u;
  u.q[0] = *reinterpret_cast<const f4*>(p);        // K base+0..7
  u.q[1] = *reinterpret_cast<const f4*>(p + 16);   // K base+16..23
  return u.v;
}

// B fragment (32x16, 16-bit): lane = N%16; lanes 0-15 hold K [0..7] and
// [16..23], lanes 16-31 hold K [8..15] and [24..31]. Source must be [n][k] in LDS.
__device__ __forceinline__ bf16x16 ld_frag_b(const bf16* base, int n0,
                                             int stride, int k0) {
  const int lane = threadIdx.x & 31;
  const bf16* p = base + (size_t)(n0 + (lane & 15)) * stride + k0 + (lane >> 4) * 8;
  FragU u;
  u.q[0] = *reinterpret_cast<const f4*>(p);
  u.q[1] = *reinterpret_cast<const f4*>(p + 16);
  return u.v;
}

// ---------------------------------------------------------------------------
// Generic GEMM: C[M,N] = A[M,K] * B[K,N], f32 in/out, bf16 WMMA compute.
// Block tile 128x128x64, 256 threads (8 waves), wave tile 64x32 (4x2 WMMA).
// Double-buffered LDS: next tile's global loads are issued before the current
// tile's WMMA loop so HBM latency hides under the matrix pipe.
// ---------------------------------------------------------------------------
#define GBK  64
#define GSTR 72   // 64 + 8 pad (bf16 elems, keeps 16B alignment)

__global__ __launch_bounds__(256)
void gemm_f32_bf16wmma(const float* __restrict__ A, const float* __restrict__ B,
                       float* __restrict__ C, int M, int N, int K)
{
  __shared__ __align__(16) bf16 sA [2][128 * GSTR];   // [m][k]
  __shared__ __align__(16) bf16 sBt[2][128 * GSTR];   // [n][k] (transposed)

  const int tid = threadIdx.x;
  const int m0 = blockIdx.y * 128;
  const int n0 = blockIdx.x * 128;
  const int wave = tid >> 5;
  const int wm = (wave & 1) * 64;
  const int wn = (wave >> 1) * 32;

  floatx8 acc[4][2];
#pragma unroll
  for (int i = 0; i < 4; ++i)
#pragma unroll
    for (int j = 0; j < 2; ++j) acc[i][j] = {};

  f4 ra[8], rb[8];   // staging registers: 8 overlapped b128 loads each

  auto load_tile = [&](int k0) {
#pragma unroll
    for (int i = 0; i < 8; ++i) {           // A tile 128x64: 2048 f4 slots
      int lin = tid + i * 256;
      int row = lin >> 4;
      int c   = (lin & 15) * 4;
      ra[i] = *reinterpret_cast<const f4*>(A + (size_t)(m0 + row) * K + k0 + c);
    }
#pragma unroll
    for (int i = 0; i < 8; ++i) {           // B tile 64x128
      int lin = tid + i * 256;
      int kk = lin >> 5;
      int nc = (lin & 31) * 4;
      rb[i] = *reinterpret_cast<const f4*>(B + (size_t)(k0 + kk) * N + n0 + nc);
    }
  };

  auto store_tile = [&](int buf) {
#pragma unroll
    for (int i = 0; i < 8; ++i) {
      int lin = tid + i * 256;
      int row = lin >> 4;
      int c   = (lin & 15) * 4;
      bf16* d = &sA[buf][row * GSTR + c];
      d[0] = (bf16)ra[i].x; d[1] = (bf16)ra[i].y;
      d[2] = (bf16)ra[i].z; d[3] = (bf16)ra[i].w;
    }
#pragma unroll
    for (int i = 0; i < 8; ++i) {
      int lin = tid + i * 256;
      int kk = lin >> 5;
      int nc = (lin & 31) * 4;
      sBt[buf][(nc + 0) * GSTR + kk] = (bf16)rb[i].x;
      sBt[buf][(nc + 1) * GSTR + kk] = (bf16)rb[i].y;
      sBt[buf][(nc + 2) * GSTR + kk] = (bf16)rb[i].z;
      sBt[buf][(nc + 3) * GSTR + kk] = (bf16)rb[i].w;
    }
  };

  const int ntiles = K / GBK;
  load_tile(0);
  store_tile(0);
  __syncthreads();

  for (int kt = 0; kt < ntiles; ++kt) {
    if (kt + 1 < ntiles) load_tile((kt + 1) * GBK);   // loads in flight now

    const bf16* cA = &sA[kt & 1][0];
    const bf16* cB = &sBt[kt & 1][0];
#pragma unroll
    for (int ks = 0; ks < GBK; ks += 32) {
      bf16x16 af[4], bfv[2];
#pragma unroll
      for (int i = 0; i < 4; ++i) af[i] = ld_frag_a(cA, wm + i * 16, GSTR, ks);
#pragma unroll
      for (int j = 0; j < 2; ++j) bfv[j] = ld_frag_b(cB, wn + j * 16, GSTR, ks);
#pragma unroll
      for (int i = 0; i < 4; ++i)
#pragma unroll
        for (int j = 0; j < 2; ++j)
          acc[i][j] = wmma_bf16(af[i], bfv[j], acc[i][j]);
    }

    if (kt + 1 < ntiles) store_tile((kt + 1) & 1);    // other buffer: no race
    __syncthreads();
  }

  // C/D layout: col = lane%16, rows = (lane/16)*8 + vgpr
  const int lane = tid & 31;
  const int cn = lane & 15;
  const int rg = (lane >> 4) * 8;
#pragma unroll
  for (int i = 0; i < 4; ++i)
#pragma unroll
    for (int j = 0; j < 2; ++j) {
      size_t base = (size_t)(m0 + wm + i * 16 + rg) * N + n0 + wn + j * 16 + cn;
#pragma unroll
      for (int r = 0; r < 8; ++r)
        C[base + (size_t)r * N] = acc[i][j][r];
    }
}

// ---------------------------------------------------------------------------
// RoPE (in-place, f32): x layout [2*1024 rows][nheads][128]
// ---------------------------------------------------------------------------
__global__ void rope_kernel(float* __restrict__ x, int nheads, int total)
{
  int tid = blockIdx.x * blockDim.x + threadIdx.x;
  if (tid >= total) return;
  int i   = tid & 63;
  int h   = (tid >> 6) % nheads;
  int row = tid / (64 * nheads);
  int s   = row & 1023;
  float fe  = (float)(2 * i) * (1.0f / 128.0f);
  float inv = __expf(-fe * 9.210340371976184f);   // ln(10000)
  float ang = (float)s * inv;
  float sn, c;
  __sincosf(ang, &sn, &c);
  size_t base = ((size_t)row * nheads + h) * 128;
  float x1 = x[base + i], x2 = x[base + 64 + i];
  x[base + i]      = x1 * c - x2 * sn;
  x[base + 64 + i] = x2 * c + x1 * sn;
}

// ---------------------------------------------------------------------------
// Flash attention: block = (b, q-head, 64-row q tile), 4 waves x 16 rows.
// K/V tiles of 128 keys, online softmax, all matmuls via WMMA bf16.
// ---------------------------------------------------------------------------
#define ASTR 136   // 128 + 8 pad

__global__ __launch_bounds__(128)
void attn_fwd(const float* __restrict__ Qp, const float* __restrict__ Kp,
              const float* __restrict__ Vp, const float* __restrict__ mask,
              float* __restrict__ O)
{
  const int S = 1024, DH = 128, NQH = 32, NKVH = 8;
  __shared__ __align__(16) bf16 sQ [64  * ASTR];     // [q][d], pre-scaled
  __shared__ __align__(16) bf16 sK [128 * ASTR];     // [key][d]  (B-frag ready)
  __shared__ __align__(16) bf16 sVt[128 * ASTR];     // [d][key]  (B-frag ready)
  __shared__ __align__(16) bf16 sP [4][16 * ASTR];   // per-wave P scratch

  const int blk = blockIdx.x;
  const int mt  = blk & 15;
  const int h   = (blk >> 4) & 31;
  const int b   = blk >> 9;
  const int kvh = h >> 2;                 // g = 32/8 = 4
  const int tid = threadIdx.x;
  const int lane = tid & 31;
  const int wave = tid >> 5;
  const int cn = lane & 15;
  const int rg = (lane >> 4) * 8;
  const int q0 = mt * 64;
  const float scale = 0.08838834764831845f;   // 1/sqrt(128), folded into Q

  // stage Q tile 64x128 in chunks of 4 overlapped b128 loads
#pragma unroll
  for (int i = 0; i < 4; ++i) {
    f4 qr[4];
#pragma unroll
    for (int j = 0; j < 4; ++j) {
      int lin = tid + (i * 4 + j) * 128;
      int row = lin >> 5;
      int c   = (lin & 31) * 4;
      qr[j] = *reinterpret_cast<const f4*>(
          Qp + (((size_t)(b * S + q0 + row) * NQH) + h) * DH + c);
    }
#pragma unroll
    for (int j = 0; j < 4; ++j) {
      int lin = tid + (i * 4 + j) * 128;
      int row = lin >> 5;
      int c   = (lin & 31) * 4;
      bf16* d = sQ + row * ASTR + c;
      d[0] = (bf16)(qr[j].x * scale); d[1] = (bf16)(qr[j].y * scale);
      d[2] = (bf16)(qr[j].z * scale); d[3] = (bf16)(qr[j].w * scale);
    }
  }

  float row_m[8], row_l[8];
  floatx8 accO[8];
#pragma unroll
  for (int r = 0; r < 8; ++r) { row_m[r] = -3.0e30f; row_l[r] = 0.0f; }
#pragma unroll
  for (int dn = 0; dn < 8; ++dn) accO[dn] = {};

  for (int kt = 0; kt < 8; ++kt) {
    const int ks0 = kt * 128;
    // stage K [key][d] and V^T [d][key]: chunks of 8 overlapped b128 loads
#pragma unroll
    for (int i = 0; i < 8; ++i) {
      f4 kr[4], vr[4];
#pragma unroll
      for (int j = 0; j < 4; ++j) {
        int lin = tid + (i * 4 + j) * 128;
        int row = lin >> 5;              // key 0..127
        int c   = (lin & 31) * 4;        // d
        size_t gb = (((size_t)(b * S + ks0 + row) * NKVH) + kvh) * DH + c;
        kr[j] = *reinterpret_cast<const f4*>(Kp + gb);
        vr[j] = *reinterpret_cast<const f4*>(Vp + gb);
      }
#pragma unroll
      for (int j = 0; j < 4; ++j) {
        int lin = tid + (i * 4 + j) * 128;
        int row = lin >> 5;
        int c   = (lin & 31) * 4;
        bf16* d = sK + row * ASTR + c;
        d[0] = (bf16)kr[j].x; d[1] = (bf16)kr[j].y;
        d[2] = (bf16)kr[j].z; d[3] = (bf16)kr[j].w;
        sVt[(c + 0) * ASTR + row] = (bf16)vr[j].x;
        sVt[(c + 1) * ASTR + row] = (bf16)vr[j].y;
        sVt[(c + 2) * ASTR + row] = (bf16)vr[j].z;
        sVt[(c + 3) * ASTR + row] = (bf16)vr[j].w;
      }
    }
    __syncthreads();

    // global_prefetch_b8 hints for next K/V tile (L2/WGP$ warm-up)
    if (kt + 1 < 8) {
      size_t nb = (((size_t)(b * S + ks0 + 128 + (tid >> 1)) * NKVH) + kvh) * DH
                  + (tid & 1) * 64;
      __builtin_prefetch(Kp + nb, 0, 1);
      __builtin_prefetch(Vp + nb, 0, 1);
    }

    // S = (Q*scale) K^T : 8 col-tiles x 4 k-steps of WMMA
    bf16x16 qf[4];
#pragma unroll
    for (int kk = 0; kk < 4; ++kk) qf[kk] = ld_frag_a(sQ, wave * 16, ASTR, kk * 32);
    floatx8 sc[8];
#pragma unroll
    for (int kn = 0; kn < 8; ++kn) {
      sc[kn] = {};
#pragma unroll
      for (int kk = 0; kk < 4; ++kk)
        sc[kn] = wmma_bf16(qf[kk], ld_frag_b(sK, kn * 16, ASTR, kk * 32), sc[kn]);
    }

    // mask + online softmax (row stats across the 16-lane column group)
#pragma unroll
    for (int r = 0; r < 8; ++r) {
      const int qrow = q0 + wave * 16 + rg + r;
      float mx = -3.0e30f;
#pragma unroll
      for (int kn = 0; kn < 8; ++kn) {
        int kcol = ks0 + kn * 16 + cn;
        float mval = mask[((size_t)b * S + qrow) * S + kcol];
        float sv = (mval != 0.0f) ? sc[kn][r] : -1.0e9f;
        sc[kn][r] = sv;
        mx = fmaxf(mx, sv);
      }
#pragma unroll
      for (int off = 1; off < 16; off <<= 1)
        mx = fmaxf(mx, __shfl_xor(mx, off, 32));
      float m_new = fmaxf(row_m[r], mx);
      float corr  = __expf(row_m[r] - m_new);
      float lsum  = 0.0f;
#pragma unroll
      for (int kn = 0; kn < 8; ++kn) {
        float p = __expf(sc[kn][r] - m_new);
        sc[kn][r] = p;
        lsum += p;
      }
#pragma unroll
      for (int off = 1; off < 16; off <<= 1)
        lsum += __shfl_xor(lsum, off, 32);
      row_l[r] = row_l[r] * corr + lsum;
      row_m[r] = m_new;
#pragma unroll
      for (int dn = 0; dn < 8; ++dn) accO[dn][r] *= corr;
    }

    // P (accumulator layout) -> LDS -> A-fragment layout
    bf16* myP = &sP[wave][0];
#pragma unroll
    for (int kn = 0; kn < 8; ++kn)
#pragma unroll
      for (int r = 0; r < 8; ++r)
        myP[(rg + r) * ASTR + kn * 16 + cn] = (bf16)sc[kn][r];
    __syncthreads();

    // O += P V
    bf16x16 pf[4];
#pragma unroll
    for (int kk = 0; kk < 4; ++kk) pf[kk] = ld_frag_a(myP, 0, ASTR, kk * 32);
#pragma unroll
    for (int dn = 0; dn < 8; ++dn)
#pragma unroll
      for (int kk = 0; kk < 4; ++kk)
        accO[dn] = wmma_bf16(pf[kk], ld_frag_b(sVt, dn * 16, ASTR, kk * 32), accO[dn]);
    __syncthreads();
  }

  // write O[b, s, h*128+d] = acc / row_sum
#pragma unroll
  for (int dn = 0; dn < 8; ++dn)
#pragma unroll
    for (int r = 0; r < 8; ++r) {
      int row = q0 + wave * 16 + rg + r;
      int col = dn * 16 + cn;
      O[(((size_t)(b * S + row) * NQH) + h) * DH + col] = accO[dn][r] / row_l[r];
    }
}

// ---------------------------------------------------------------------------
// Host-side launcher
// ---------------------------------------------------------------------------
extern "C" void kernel_launch(void* const* d_in, const int* in_sizes, int n_in,
                              void* d_out, int out_size, void* d_ws, size_t ws_size,
                              hipStream_t stream) {
  const float* query = (const float*)d_in[0];   // [2,1024,4096]
  const float* key   = (const float*)d_in[1];
  const float* value = (const float*)d_in[2];
  const float* mask  = (const float*)d_in[3];   // [2,1024,1024]
  const float* wq    = (const float*)d_in[4];   // [4096,4096]
  const float* wk    = (const float*)d_in[5];   // [4096,1024]
  const float* wv    = (const float*)d_in[6];   // [4096,1024]
  const float* wo    = (const float*)d_in[7];   // [4096,4096]
  float* out = (float*)d_out;

  float* ws = (float*)d_ws;
  float* qp = ws;                                // 2048*4096
  float* kp = qp + (size_t)2048 * 4096;          // 2048*1024
  float* vp = kp + (size_t)2048 * 1024;          // 2048*1024
  float* ao = vp + (size_t)2048 * 1024;          // 2048*4096

  // projections
  gemm_f32_bf16wmma<<<dim3(32, 16), 256, 0, stream>>>(query, wq, qp, 2048, 4096, 4096);
  gemm_f32_bf16wmma<<<dim3( 8, 16), 256, 0, stream>>>(key,   wk, kp, 2048, 1024, 4096);
  gemm_f32_bf16wmma<<<dim3( 8, 16), 256, 0, stream>>>(value, wv, vp, 2048, 1024, 4096);

  // RoPE on q and k
  {
    int tq = 2048 * 32 * 64;
    rope_kernel<<<(tq + 255) / 256, 256, 0, stream>>>(qp, 32, tq);
    int tk = 2048 * 8 * 64;
    rope_kernel<<<(tk + 255) / 256, 256, 0, stream>>>(kp, 8, tk);
  }

  // flash attention: 2 batches * 32 heads * 16 q-tiles = 1024 blocks
  attn_fwd<<<1024, 128, 0, stream>>>(qp, kp, vp, mask, ao);

  // output projection
  gemm_f32_bf16wmma<<<dim3(32, 16), 256, 0, stream>>>(ao, wo, out, 2048, 4096, 4096);
}
